// TimeIPAttnProcessor_69441031241977
// MI455X (gfx1250) — matrix operand order repaired and checked by reading, not compiled, not run on hardware
//
#include <hip/hip_runtime.h>
#include <stdint.h>

// ---------------------------------------------------------------------------
// Problem constants
// ---------------------------------------------------------------------------
#define B_   8
#define S_   4096
#define D_   1280
#define H_   20
#define DH_  64
#define DC_  768
#define TT_  77     // text tokens
#define TTP_ 96     // text tokens padded (multiple of 32 for WMMA K)
#define TIP_ 4      // ip tokens
#define TIPP_ 32    // ip tokens padded
#define SMP_ 16     // sm rows padded

typedef unsigned short u16;
typedef unsigned int   u32;
typedef __attribute__((ext_vector_type(16))) __bf16 bf16x16;
typedef __attribute__((ext_vector_type(8)))  float  v8f;

union Frag32 {            // 32-byte WMMA operand fragment (16 x bf16 per lane)
    bf16x16 v;
    uint4   q[2];
};

__device__ __forceinline__ u16 f2bf(float f) {
    u32 u = __float_as_uint(f);
    u32 r = (u + 0x7FFFu + ((u >> 16) & 1u)) >> 16;   // round-to-nearest-even
    return (u16)r;
}

__device__ __forceinline__ v8f wmma_bf16(const Frag32& a, const Frag32& b, v8f c) {
    return __builtin_amdgcn_wmma_f32_16x16x32_bf16(
        false, a.v, false, b.v, (short)0, c, false, false);
}

// ---------------------------------------------------------------------------
// Workspace layout (bytes).  attn_bf aliases hidden_bf (hidden_bf is fully
// consumed by the Q-projection GEMM before attention writes attn_bf).
// ---------------------------------------------------------------------------
static constexpr size_t OFF_HIDBF  = 0;                                    // 83,886,080
static constexpr size_t OFF_WQT    = OFF_HIDBF + (size_t)B_*S_*D_*2;       //  3,276,800
static constexpr size_t OFF_WOT    = OFF_WQT   + (size_t)D_*D_*2;          //  3,276,800
static constexpr size_t OFF_QBF    = OFF_WOT   + (size_t)D_*D_*2;          // 83,886,080
static constexpr size_t OFF_KBF    = OFF_QBF   + (size_t)B_*S_*D_*2;       //  1,966,080
static constexpr size_t OFF_VT     = OFF_KBF   + (size_t)B_*H_*TTP_*DH_*2; //  1,966,080
static constexpr size_t OFF_IPK    = OFF_VT    + (size_t)B_*H_*DH_*TTP_*2; //    655,360
static constexpr size_t OFF_IPVT   = OFF_IPK   + (size_t)B_*H_*TIPP_*DH_*2;//    655,360
static constexpr size_t OFF_IPKF   = OFF_IPVT  + (size_t)B_*H_*DH_*TIPP_*2;//    163,840
static constexpr size_t OFF_SMBF   = OFF_IPKF  + (size_t)B_*H_*TIP_*DH_*4; //    327,680
static constexpr size_t OFF_ATTNBF = OFF_HIDBF;                            // alias

// ---------------------------------------------------------------------------
// Kernel 1: fp32 -> bf16 bulk convert (4 elems / thread)
// ---------------------------------------------------------------------------
__global__ __launch_bounds__(256) void k_f32_to_bf16(const float* __restrict__ x,
                                                     u16* __restrict__ y, long n4) {
    long i = (long)blockIdx.x * blockDim.x + threadIdx.x;
    if (i >= n4) return;
    float4 f = ((const float4*)x)[i];
    u32 lo = (u32)f2bf(f.x) | ((u32)f2bf(f.y) << 16);
    u32 hi = (u32)f2bf(f.z) | ((u32)f2bf(f.w) << 16);
    ((uint2*)y)[i] = make_uint2(lo, hi);
}

// ---------------------------------------------------------------------------
// Kernel 2: W (KxN, fp32) -> WT (NxK, bf16), 32x32 LDS tiles
// ---------------------------------------------------------------------------
__global__ __launch_bounds__(256) void k_transpose_bf16(const float* __restrict__ W,
                                                        u16* __restrict__ WT) {
    __shared__ float tile[32][33];
    int tx = threadIdx.x & 31, ty0 = threadIdx.x >> 5;   // 32x8 threads
    int bx = blockIdx.x, by = blockIdx.y;
#pragma unroll
    for (int i = 0; i < 4; i++) {
        int ty = ty0 + i * 8;
        tile[ty][tx] = W[(size_t)(by * 32 + ty) * D_ + bx * 32 + tx];
    }
    __syncthreads();
#pragma unroll
    for (int i = 0; i < 4; i++) {
        int ty = ty0 + i * 8;
        WT[(size_t)(bx * 32 + ty) * D_ + by * 32 + tx] = f2bf(tile[tx][ty]);
    }
}

// ---------------------------------------------------------------------------
// Kernel 3: text K/V projections -> Kbf[b][h][96][64], Vt[b][h][64][96] (bf16)
// grid = B_*12 blocks (8 token-rows each), 256 threads
// ---------------------------------------------------------------------------
__global__ __launch_bounds__(256) void k_prep_kv_text(const float* __restrict__ enc,
                                                      const float* __restrict__ Wk,
                                                      const float* __restrict__ Wv,
                                                      u16* __restrict__ Kbf,
                                                      u16* __restrict__ Vt) {
    __shared__ float sh[8][DC_];
    int b = blockIdx.x / 12;
    int t0 = (blockIdx.x % 12) * 8;
    for (int i = threadIdx.x; i < 8 * DC_; i += 256) {
        int rr = i / DC_, kk = i % DC_;
        int t = t0 + rr;
        sh[rr][kk] = (t < TT_) ? enc[((size_t)b * 81 + t) * DC_ + kk] : 0.f;
    }
    __syncthreads();
    for (int n = threadIdx.x; n < D_; n += 256) {
        float ak[8] = {0.f,0.f,0.f,0.f,0.f,0.f,0.f,0.f};
        float av[8] = {0.f,0.f,0.f,0.f,0.f,0.f,0.f,0.f};
        for (int k = 0; k < DC_; k++) {
            float wk = Wk[(size_t)k * D_ + n];
            float wv = Wv[(size_t)k * D_ + n];
#pragma unroll
            for (int rr = 0; rr < 8; rr++) {
                float x = sh[rr][k];
                ak[rr] = fmaf(x, wk, ak[rr]);
                av[rr] = fmaf(x, wv, av[rr]);
            }
        }
        int h = n >> 6, dh = n & 63;
#pragma unroll
        for (int rr = 0; rr < 8; rr++) {
            int t = t0 + rr;
            u16 kv = (t < TT_) ? f2bf(ak[rr]) : (u16)0;
            u16 vv = (t < TT_) ? f2bf(av[rr]) : (u16)0;
            Kbf[(((size_t)(b * H_ + h)) * TTP_ + t) * DH_ + dh] = kv;
            Vt [(((size_t)(b * H_ + h)) * DH_ + dh) * TTP_ + t] = vv;
        }
    }
}

// ---------------------------------------------------------------------------
// Kernel 4: ip K/V projections.  grid = B_ blocks.
// ipK[b][h][32][64] bf16, ipVt[b][h][64][32] bf16, ipKf[b][h][4][64] fp32
// ---------------------------------------------------------------------------
__global__ __launch_bounds__(256) void k_prep_ip(const float* __restrict__ enc,
                                                 const float* __restrict__ Wki,
                                                 const float* __restrict__ Wvi,
                                                 u16* __restrict__ ipK,
                                                 u16* __restrict__ ipVt,
                                                 float* __restrict__ ipKf) {
    __shared__ float sh[TIP_][DC_];
    int b = blockIdx.x;
    for (int i = threadIdx.x; i < TIP_ * DC_; i += 256) {
        int rr = i / DC_, kk = i % DC_;
        sh[rr][kk] = enc[((size_t)b * 81 + TT_ + rr) * DC_ + kk];
    }
    // zero the pad region (tokens 4..31) -- disjoint from value writes below
    for (int i = threadIdx.x; i < H_ * DH_ * (TIPP_ - TIP_); i += 256) {
        int t  = TIP_ + (i % (TIPP_ - TIP_));
        int hd = i / (TIPP_ - TIP_);
        int h = hd >> 6, dh = hd & 63;
        ipK [(((size_t)(b * H_ + h)) * TIPP_ + t) * DH_ + dh] = 0;
        ipVt[(((size_t)(b * H_ + h)) * DH_ + dh) * TIPP_ + t] = 0;
    }
    __syncthreads();
    for (int n = threadIdx.x; n < D_; n += 256) {
        float ak[TIP_] = {0.f,0.f,0.f,0.f};
        float av[TIP_] = {0.f,0.f,0.f,0.f};
        for (int k = 0; k < DC_; k++) {
            float wk = Wki[(size_t)k * D_ + n];
            float wv = Wvi[(size_t)k * D_ + n];
#pragma unroll
            for (int rr = 0; rr < TIP_; rr++) {
                float x = sh[rr][k];
                ak[rr] = fmaf(x, wk, ak[rr]);
                av[rr] = fmaf(x, wv, av[rr]);
            }
        }
        int h = n >> 6, dh = n & 63;
#pragma unroll
        for (int rr = 0; rr < TIP_; rr++) {
            ipK [(((size_t)(b * H_ + h)) * TIPP_ + rr) * DH_ + dh] = f2bf(ak[rr]);
            ipVt[(((size_t)(b * H_ + h)) * DH_ + dh) * TIPP_ + rr] = f2bf(av[rr]);
            ipKf[(((size_t)(b * H_ + h)) * TIP_  + rr) * DH_ + dh] = ak[rr];
        }
    }
}

// ---------------------------------------------------------------------------
// Kernel 5: sm = softmax(ip_k over tokens) -> smbf[b][h][16][64] (rows 4..15 = 0)
// ---------------------------------------------------------------------------
__global__ __launch_bounds__(256) void k_prep_sm(const float* __restrict__ ipKf,
                                                 u16* __restrict__ smbf) {
    int i = blockIdx.x * 256 + threadIdx.x;     // B*H*64 = 10240
    if (i >= B_ * H_ * DH_) return;
    int dh = i & 63, bh = i >> 6;
    float v[TIP_], mx = -1e30f;
#pragma unroll
    for (int t = 0; t < TIP_; t++) {
        v[t] = ipKf[((size_t)bh * TIP_ + t) * DH_ + dh];
        mx = fmaxf(mx, v[t]);
    }
    float s = 0.f;
#pragma unroll
    for (int t = 0; t < TIP_; t++) { v[t] = __expf(v[t] - mx); s += v[t]; }
    float rs = 1.f / s;
#pragma unroll
    for (int t = 0; t < TIP_; t++)
        smbf[((size_t)bh * SMP_ + t) * DH_ + dh] = f2bf(v[t] * rs);
#pragma unroll
    for (int t = TIP_; t < SMP_; t++)
        smbf[((size_t)bh * SMP_ + t) * DH_ + dh] = 0;
}

// ---------------------------------------------------------------------------
// Kernel 6: bf16 WMMA GEMM  C[M][1280] = A[M][1280] x W[1280][1280]
// A row-major bf16, WT = W^T row-major bf16.  256 thr = 8 waves (4m x 2n),
// block tile 128x128, each wave 32x64 (2x4 WMMA tiles), K-step 32.
// mode 0: store bf16 -> Obf.   mode 1: f32 + bias + residual -> Of.
// ---------------------------------------------------------------------------
__global__ __launch_bounds__(256) void k_gemm_bf16(const u16* __restrict__ A,
                                                   const u16* __restrict__ WT,
                                                   int mode,
                                                   u16* __restrict__ Obf,
                                                   float* __restrict__ Of,
                                                   const float* __restrict__ bias,
                                                   const float* __restrict__ resid) {
    const int K = D_;
    int lane = threadIdx.x & 31, w = threadIdx.x >> 5;
    int half = lane >> 4, r = lane & 15;
    int wm = w >> 1, wn = w & 1;
    size_t m0 = (size_t)blockIdx.y * 128 + wm * 32;
    int    n0 = blockIdx.x * 128 + wn * 64;

    v8f acc[2][4];
#pragma unroll
    for (int mt = 0; mt < 2; mt++)
#pragma unroll
        for (int nt = 0; nt < 4; nt++) acc[mt][nt] = {};

    for (int k0 = 0; k0 < K; k0 += 32) {
        Frag32 a[2], bf[4];
#pragma unroll
        for (int mt = 0; mt < 2; mt++) {
            const u16* p = A + (m0 + mt * 16 + r) * (size_t)K + k0 + half * 16;
            a[mt].q[0] = *(const uint4*)p;
            a[mt].q[1] = *(const uint4*)(p + 8);
        }
#pragma unroll
        for (int nt = 0; nt < 4; nt++) {
            const u16* p = WT + (size_t)(n0 + nt * 16 + r) * K + k0 + half * 16;
            bf[nt].q[0] = *(const uint4*)p;
            bf[nt].q[1] = *(const uint4*)(p + 8);
        }
#pragma unroll
        for (int mt = 0; mt < 2; mt++)
#pragma unroll
            for (int nt = 0; nt < 4; nt++)
                acc[mt][nt] = wmma_bf16(a[mt], bf[nt], acc[mt][nt]);
    }

#pragma unroll
    for (int mt = 0; mt < 2; mt++)
#pragma unroll
        for (int nt = 0; nt < 4; nt++) {
            int col = n0 + nt * 16 + r;
#pragma unroll
            for (int j = 0; j < 8; j++) {
                size_t row = m0 + mt * 16 + j + 8 * half;
                if (mode == 0) {
                    Obf[row * D_ + col] = f2bf(acc[mt][nt][j]);
                } else {
                    Of[row * D_ + col] =
                        acc[mt][nt][j] + bias[col] + resid[row * D_ + col];
                }
            }
        }
}

// ---------------------------------------------------------------------------
// Kernel 7: fused dual attention + attn_map.
// 128 thr = 4 waves; each wave owns one 16-row Q tile of one (b,h).
// txt: scores 16x96 (masked >=77), softmax, P·V (K=96).
// ip : scores 16x32 (masked >=4),  softmax, P·V (K=32).
// map: Q @ sm^T (16x16, store cols 0..3, f32).
// out: txt/l_txt + ip/l_ip -> bf16 attn_bf[b*S+s][h*64+dh].
// ---------------------------------------------------------------------------
__global__ __launch_bounds__(128) void k_attention(const u16* __restrict__ Qbf,
                                                   const u16* __restrict__ Kbf,
                                                   const u16* __restrict__ Vt,
                                                   const u16* __restrict__ ipK,
                                                   const u16* __restrict__ ipVt,
                                                   const u16* __restrict__ smbf,
                                                   u16* __restrict__ attn_bf,
                                                   float* __restrict__ outmap) {
    __shared__ u16 Pbuf[4][16 * TTP_];           // per-wave P tile (12 KB total)
    int wid = threadIdx.x >> 5;
    int lane = threadIdx.x & 31;
    int half = lane >> 4, r = lane & 15;
    u16* P = &Pbuf[wid][0];

    int tile = blockIdx.x * 4 + wid;             // 40960 tiles total
    int st = tile & 255;                         // S/16 = 256
    int bh = tile >> 8;
    int b = bh / H_, h = bh % H_;
    int s0 = st * 16;

    // ---- Q A-fragments (shared by txt scores, ip scores, attn_map) ----
    const u16* Qrow = Qbf + ((size_t)b * S_ + s0 + r) * D_ + h * DH_;
    Frag32 qa[2];
#pragma unroll
    for (int ks = 0; ks < 2; ks++) {
        const u16* p = Qrow + ks * 32 + half * 16;
        qa[ks].q[0] = *(const uint4*)p;
        qa[ks].q[1] = *(const uint4*)(p + 8);
    }

    // ================= text attention =================
    const u16* Kb = Kbf + ((size_t)(b * H_ + h)) * TTP_ * DH_;
    v8f sc[6];
#pragma unroll
    for (int t = 0; t < 6; t++) {
        v8f c = {};
#pragma unroll
        for (int ks = 0; ks < 2; ks++) {
            Frag32 kb;
            const u16* p = Kb + (size_t)(t * 16 + r) * DH_ + ks * 32 + half * 16;
            kb.q[0] = *(const uint4*)p;
            kb.q[1] = *(const uint4*)(p + 8);
            c = wmma_bf16(qa[ks], kb, c);
        }
        int col = t * 16 + r;
        bool ok = col < TT_;
#pragma unroll
        for (int j = 0; j < 8; j++)
            c[j] = ok ? c[j] * 0.125f : -1e30f;
        sc[t] = c;
    }
    // rowwise softmax (rows j+8*half live in one 16-lane half)
    float m[8], l[8];
#pragma unroll
    for (int j = 0; j < 8; j++) {
        float mm = sc[0][j];
#pragma unroll
        for (int t = 1; t < 6; t++) mm = fmaxf(mm, sc[t][j]);
        for (int d = 1; d < 16; d <<= 1) mm = fmaxf(mm, __shfl_xor(mm, d, 32));
        m[j] = mm;
    }
#pragma unroll
    for (int t = 0; t < 6; t++)
#pragma unroll
        for (int j = 0; j < 8; j++) sc[t][j] = __expf(sc[t][j] - m[j]);
#pragma unroll
    for (int j = 0; j < 8; j++) {
        float ss = 0.f;
#pragma unroll
        for (int t = 0; t < 6; t++) ss += sc[t][j];
        for (int d = 1; d < 16; d <<= 1) ss += __shfl_xor(ss, d, 32);
        l[j] = 1.f / ss;
    }
    // stash P (bf16) in LDS, re-fragment as A for P·V
#pragma unroll
    for (int t = 0; t < 6; t++)
#pragma unroll
        for (int j = 0; j < 8; j++)
            P[(j + 8 * half) * TTP_ + t * 16 + r] = f2bf(sc[t][j]);

    const u16* Vb = Vt + ((size_t)(b * H_ + h)) * DH_ * TTP_;
    v8f ov[4];
#pragma unroll
    for (int nt = 0; nt < 4; nt++) ov[nt] = {};
#pragma unroll
    for (int ks = 0; ks < 3; ks++) {
        Frag32 pa;
        const u16* pp = P + r * TTP_ + ks * 32 + half * 16;
        pa.q[0] = *(const uint4*)pp;
        pa.q[1] = *(const uint4*)(pp + 8);
#pragma unroll
        for (int nt = 0; nt < 4; nt++) {
            Frag32 vb;
            const u16* p = Vb + (size_t)(nt * 16 + r) * TTP_ + ks * 32 + half * 16;
            vb.q[0] = *(const uint4*)p;
            vb.q[1] = *(const uint4*)(p + 8);
            ov[nt] = wmma_bf16(pa, vb, ov[nt]);
        }
    }
#pragma unroll
    for (int nt = 0; nt < 4; nt++)
#pragma unroll
        for (int j = 0; j < 8; j++) ov[nt][j] *= l[j];

    // ================= ip attention =================
    const u16* Kib = ipK + ((size_t)(b * H_ + h)) * TIPP_ * DH_;
    v8f si[2];
#pragma unroll
    for (int t = 0; t < 2; t++) {
        v8f c = {};
#pragma unroll
        for (int ks = 0; ks < 2; ks++) {
            Frag32 kb;
            const u16* p = Kib + (size_t)(t * 16 + r) * DH_ + ks * 32 + half * 16;
            kb.q[0] = *(const uint4*)p;
            kb.q[1] = *(const uint4*)(p + 8);
            c = wmma_bf16(qa[ks], kb, c);
        }
        int col = t * 16 + r;
        bool ok = col < TIP_;
#pragma unroll
        for (int j = 0; j < 8; j++)
            c[j] = ok ? c[j] * 0.125f : -1e30f;
        si[t] = c;
    }
    float mi[8], li[8];
#pragma unroll
    for (int j = 0; j < 8; j++) {
        float mm = fmaxf(si[0][j], si[1][j]);
        for (int d = 1; d < 16; d <<= 1) mm = fmaxf(mm, __shfl_xor(mm, d, 32));
        mi[j] = mm;
    }
#pragma unroll
    for (int t = 0; t < 2; t++)
#pragma unroll
        for (int j = 0; j < 8; j++) si[t][j] = __expf(si[t][j] - mi[j]);
#pragma unroll
    for (int j = 0; j < 8; j++) {
        float ss = si[0][j] + si[1][j];
        for (int d = 1; d < 16; d <<= 1) ss += __shfl_xor(ss, d, 32);
        li[j] = 1.f / ss;
    }
#pragma unroll
    for (int t = 0; t < 2; t++)
#pragma unroll
        for (int j = 0; j < 8; j++)
            P[(j + 8 * half) * TIPP_ + t * 16 + r] = f2bf(si[t][j]);

    const u16* Vib = ipVt + ((size_t)(b * H_ + h)) * DH_ * TIPP_;
    v8f oi[4];
    {
        Frag32 pa;
        const u16* pp = P + r * TIPP_ + half * 16;
        pa.q[0] = *(const uint4*)pp;
        pa.q[1] = *(const uint4*)(pp + 8);
#pragma unroll
        for (int nt = 0; nt < 4; nt++) {
            Frag32 vb;
            const u16* p = Vib + (size_t)(nt * 16 + r) * TIPP_ + half * 16;
            vb.q[0] = *(const uint4*)p;
            vb.q[1] = *(const uint4*)(p + 8);
            v8f c = {};
            oi[nt] = wmma_bf16(pa, vb, c);
        }
    }
#pragma unroll
    for (int nt = 0; nt < 4; nt++)
#pragma unroll
        for (int j = 0; j < 8; j++) oi[nt][j] *= li[j];

    // ================= attn_map = Q @ sm^T (no scale, no softmax) ==========
    {
        const u16* Sb = smbf + ((size_t)(b * H_ + h)) * SMP_ * DH_;
        v8f cm = {};
#pragma unroll
        for (int ks = 0; ks < 2; ks++) {
            Frag32 sb;
            const u16* p = Sb + (size_t)r * DH_ + ks * 32 + half * 16;
            sb.q[0] = *(const uint4*)p;
            sb.q[1] = *(const uint4*)(p + 8);
            cm = wmma_bf16(qa[ks], sb, cm);
        }
        if (r < TIP_) {
#pragma unroll
            for (int j = 0; j < 8; j++) {
                size_t s = (size_t)s0 + j + 8 * half;
                outmap[(((size_t)(b * H_ + h)) * S_ + s) * TIP_ + r] = cm[j];
            }
        }
    }

    // ================= combine txt + ip -> bf16 =================
#pragma unroll
    for (int nt = 0; nt < 4; nt++)
#pragma unroll
        for (int j = 0; j < 8; j++) {
            float o = ov[nt][j] + oi[nt][j];
            size_t row = (size_t)b * S_ + s0 + j + 8 * half;
            attn_bf[row * D_ + h * DH_ + nt * 16 + r] = f2bf(o);
        }
}

// ---------------------------------------------------------------------------
// Host-side orchestration
// ---------------------------------------------------------------------------
extern "C" void kernel_launch(void* const* d_in, const int* in_sizes, int n_in,
                              void* d_out, int out_size, void* d_ws, size_t ws_size,
                              hipStream_t stream) {
    (void)in_sizes; (void)n_in; (void)out_size; (void)ws_size;
    const float* hidden = (const float*)d_in[0];   // (8,4096,1280)
    const float* enc    = (const float*)d_in[1];   // (8,81,768)
    const float* Wq     = (const float*)d_in[2];
    const float* Wk     = (const float*)d_in[3];
    const float* Wv     = (const float*)d_in[4];
    const float* Wk_ip  = (const float*)d_in[5];
    const float* Wv_ip  = (const float*)d_in[6];
    const float* Wo     = (const float*)d_in[7];
    const float* bo     = (const float*)d_in[8];

    char* ws = (char*)d_ws;
    u16*   hid_bf  = (u16*)  (ws + OFF_HIDBF);
    u16*   WqT     = (u16*)  (ws + OFF_WQT);
    u16*   WoT     = (u16*)  (ws + OFF_WOT);
    u16*   Qbf     = (u16*)  (ws + OFF_QBF);
    u16*   Kbf     = (u16*)  (ws + OFF_KBF);
    u16*   Vt      = (u16*)  (ws + OFF_VT);
    u16*   ipK     = (u16*)  (ws + OFF_IPK);
    u16*   ipVt    = (u16*)  (ws + OFF_IPVT);
    float* ipKf    = (float*)(ws + OFF_IPKF);
    u16*   smbf    = (u16*)  (ws + OFF_SMBF);
    u16*   attn_bf = (u16*)  (ws + OFF_ATTNBF);    // aliases hid_bf (safe)

    float* out_f   = (float*)d_out;                            // (B,S,D)
    float* out_map = (float*)d_out + (size_t)B_ * S_ * D_;     // (B,H,S,4)

    // 1) hidden -> bf16
    long n4 = (long)B_ * S_ * D_ / 4;
    k_f32_to_bf16<<<(unsigned)((n4 + 255) / 256), 256, 0, stream>>>(hidden, hid_bf, n4);

    // 2) Wq^T, Wo^T -> bf16
    k_transpose_bf16<<<dim3(D_ / 32, D_ / 32), 256, 0, stream>>>(Wq, WqT);
    k_transpose_bf16<<<dim3(D_ / 32, D_ / 32), 256, 0, stream>>>(Wo, WoT);

    // 3-5) K/V/ip/sm prep
    k_prep_kv_text<<<B_ * 12, 256, 0, stream>>>(enc, Wk, Wv, Kbf, Vt);
    k_prep_ip<<<B_, 256, 0, stream>>>(enc, Wk_ip, Wv_ip, ipK, ipVt, ipKf);
    k_prep_sm<<<(B_ * H_ * DH_ + 255) / 256, 256, 0, stream>>>(ipKf, smbf);

    // 6) Q = hidden @ Wq   (WMMA, bf16 out)
    k_gemm_bf16<<<dim3(D_ / 128, (B_ * S_) / 128), 256, 0, stream>>>(
        hid_bf, WqT, 0, Qbf, nullptr, nullptr, nullptr);

    // 7) fused attention (txt + ip + attn_map)
    k_attention<<<(B_ * H_ * (S_ / 16)) / 4, 128, 0, stream>>>(
        Qbf, Kbf, Vt, ipK, ipVt, smbf, attn_bf, out_map);

    // 8) out = attn @ Wo + bo + residual  (WMMA, f32 out)
    k_gemm_bf16<<<dim3(D_ / 128, (B_ * S_) / 128), 256, 0, stream>>>(
        attn_bf, WoT, 1, nullptr, out_f, bo, hidden);
}
